// RAFT_16295105921084
// MI455X (gfx1250) — matrix operand order repaired and checked by reading, not compiled, not run on hardware
//
#include <hip/hip_runtime.h>
#include <hip/hip_bf16.h>
#include <math.h>

typedef __attribute__((ext_vector_type(16))) _Float16 v16h;
typedef __attribute__((ext_vector_type(8)))  _Float16 v8h;
typedef __attribute__((ext_vector_type(8)))  float    v8f;

// Problem constants (from setup_inputs)
static constexpr int R_B = 4;
static constexpr int R_H = 48;
static constexpr int R_W = 64;
static constexpr int R_P = R_B * R_H * R_W;   // 12288 pixels, multiple of 64
static constexpr int RADIUS = 4;
static constexpr int KLU = (2 * RADIUS + 1) * (2 * RADIUS + 1); // 81
static constexpr int CORR_C = 4 * KLU;                          // 324
static constexpr int CORR_P = 352;                              // padded stride (%32==0)

// ---------------------------------------------------------------------------
// LDS/async implicit-GEMM conv (NHWC, stride 1, SAME), WMMA f16 -> f32.
// Block = 4 waves. Wave w computes M tile (blockIdx.x*4+w)*16 (16 pixels);
// all 4 waves share the N range blockIdx.y*64 (NT=4 tiles of 16 channels).
// Per 32-wide K chunk the block async-copies the 64x32-half weight tile
// (transposed layout [Cout, Kpad]) into LDS with global_load_async_to_lds_b128
// (ASYNCcnt) and each wave reads its B fragments from LDS (ds_load_b128).
// Fragment layouts used here (wave32, contiguous k order for A/B):
//   A 16x32 f16: lane M = lane&15; run0 elems 0..7  K=k0+(lane>>4)*16+e,
//                run1 elems 8..15 K=k0+(lane>>4)*16+8+e  (runs tap-contiguous
//                because CINP % 8 == 0)
//   B 32x16 f16: lane N = lane&15; elem e -> K = k0+(lane>>4)*16+e
//   C/D: elem r -> M = r + 8*(lane>>4), N = lane&15
// ---------------------------------------------------------------------------
template<int CINP, int KHH, int KWW>
__global__ __launch_bounds__(128)
void conv_wmma_a(const int* __restrict__ itersp, int iter,
                 const _Float16* __restrict__ in,   // [P, CINP] (pads zeroed)
                 const _Float16* __restrict__ wtT,  // [Cout, Kpad] transposed, zero padded
                 const float*    __restrict__ bias,
                 _Float16* __restrict__ out16, float* __restrict__ out32,
                 int Cout, int outStride, int outOff,
                 int act /*0 none,1 relu,2 sigmoid,3 tanh*/, float scale)
{
    if (iter >= *itersp) return;               // uniform across block
    constexpr int Kpad = KHH * KWW * CINP;
    constexpr int NT = 4;
    static_assert(Kpad % 32 == 0, "Kpad must be a multiple of 32");
    static_assert(CINP % 8 == 0, "CINP must be a multiple of 8");

    __shared__ _Float16 ldsB[64 * 32] __attribute__((aligned(128)));

    const int tid  = threadIdx.x;
    const int wv   = tid >> 5;
    const int lane = tid & 31;
    const int hs = lane >> 4;
    const int lm = lane & 15;
    const int m0 = (blockIdx.x * 4 + wv) * 16;
    const int n0 = blockIdx.y * 64;

    const int m  = m0 + lm;
    const int b  = m / (R_H * R_W);
    const int rr = m % (R_H * R_W);
    const int y  = rr / R_W;
    const int x  = rr % R_W;

    // this thread's two 16B staging segments of the 4KB weight tile
    // segment s covers row = s>>2 (0..63), col halves = (s&3)*8
    const int s0 = tid, s1 = tid + 128;
    int ng0 = n0 + (s0 >> 2); ng0 = (ng0 < Cout) ? ng0 : (Cout - 1);
    int ng1 = n0 + (s1 >> 2); ng1 = (ng1 < Cout) ? ng1 : (Cout - 1);
    const int col0 = (s0 & 3) * 8, col1 = (s1 & 3) * 8;
    const unsigned ldsBase = (unsigned)(uintptr_t)ldsB;
    const unsigned lo0 = ldsBase + (unsigned)s0 * 16u;
    const unsigned lo1 = ldsBase + (unsigned)s1 * 16u;

    const v8f vzero = {0.f, 0.f, 0.f, 0.f, 0.f, 0.f, 0.f, 0.f};
    v8f acc[NT];
#pragma unroll
    for (int t = 0; t < NT; ++t) acc[t] = vzero;

    for (int k0 = 0; k0 < Kpad; k0 += 32) {
        __syncthreads();   // previous chunk's LDS reads complete
        {
            unsigned long long ga0 =
                (unsigned long long)(uintptr_t)(wtT + (size_t)ng0 * Kpad + k0 + col0);
            unsigned long long ga1 =
                (unsigned long long)(uintptr_t)(wtT + (size_t)ng1 * Kpad + k0 + col1);
            asm volatile("global_load_async_to_lds_b128 %0, %1, off"
                         :: "v"(lo0), "v"(ga0) : "memory");
            asm volatile("global_load_async_to_lds_b128 %0, %1, off"
                         :: "v"(lo1), "v"(ga1) : "memory");
            asm volatile("s_wait_asynccnt 0x0" ::: "memory");
        }
        __syncthreads();   // whole weight tile visible to all waves

        v16h av;
#pragma unroll
        for (int run = 0; run < 2; ++run) {
            const int kstart = k0 + hs * 16 + run * 8;
            const int tap = kstart / CINP;
            const int c   = kstart % CINP;           // multiple of 8
            const int dy  = tap / KWW - (KHH >> 1);
            const int dx  = tap % KWW - (KWW >> 1);
            const int yy = y + dy, xx = x + dx;
            v8h seg = {(_Float16)0.f, (_Float16)0.f, (_Float16)0.f, (_Float16)0.f,
                       (_Float16)0.f, (_Float16)0.f, (_Float16)0.f, (_Float16)0.f};
            if (yy >= 0 && yy < R_H && xx >= 0 && xx < R_W)
                seg = *(const v8h*)(in + (size_t)(((b * R_H + yy) * R_W) + xx) * CINP + c);
#pragma unroll
            for (int e = 0; e < 8; ++e) av[run * 8 + e] = seg[e];
        }
#pragma unroll
        for (int t = 0; t < NT; ++t) {
            const v16h bv = *(const v16h*)(ldsB + (t * 16 + lm) * 32 + hs * 16);
            acc[t] = __builtin_amdgcn_wmma_f32_16x16x32_f16(false, av, false, bv,
                                                            (short)0, acc[t], false, false);
        }
    }

#pragma unroll
    for (int t = 0; t < NT; ++t) {
        const int n = n0 + t * 16 + lm;
        if (n >= Cout) continue;
        const float bval = bias ? bias[n] : 0.f;
#pragma unroll
        for (int r = 0; r < 8; ++r) {
            const int mm = m0 + hs * 8 + r;
            float v = acc[t][r] + bval;
            if (act == 1)      v = fmaxf(v, 0.f);
            else if (act == 2) v = 1.f / (1.f + __expf(-v));
            else if (act == 3) v = tanhf(v);
            v *= scale;
            const size_t o = (size_t)mm * outStride + outOff + n;
            if (out16) out16[o] = (_Float16)v;
            if (out32) out32[o] = v;
        }
    }
}

// ---------------------------------------------------------------------------
// Scalar-gather WMMA conv (only for f1: CIN=2, 7x7, K=98). Weights [K, Cout].
// ---------------------------------------------------------------------------
template<int CIN, int KHH, int KWW>
__global__ __launch_bounds__(32)
void conv_wmma_s(const int* __restrict__ itersp, int iter,
                 const _Float16* __restrict__ in,
                 const _Float16* __restrict__ wt,
                 const float*    __restrict__ bias,
                 _Float16* __restrict__ out16,
                 int Cout, int outStride, int outOff, int act)
{
    if (iter >= *itersp) return;
    constexpr int K = KHH * KWW * CIN;
    const int lane = threadIdx.x;
    const int hs = lane >> 4;
    const int lm = lane & 15;
    const int m0 = blockIdx.x * 16;
    const int n0 = blockIdx.y * 16;

    const int m  = m0 + lm;
    const int b  = m / (R_H * R_W);
    const int rr = m % (R_H * R_W);
    const int y  = rr / R_W;
    const int x  = rr % R_W;

    v8f acc = {0.f, 0.f, 0.f, 0.f, 0.f, 0.f, 0.f, 0.f};
    for (int k0 = 0; k0 < K; k0 += 32) {
        v16h av, bv;
#pragma unroll
        for (int e = 0; e < 16; ++e) {
            const int k = k0 + hs * 16 + e;
            _Float16 v = (_Float16)0.f;
            if (k < K) {
                const int tap = k / CIN;
                const int c   = k - tap * CIN;
                const int dy  = tap / KWW - (KHH >> 1);
                const int dx  = tap % KWW - (KWW >> 1);
                const int yy = y + dy, xx = x + dx;
                if (yy >= 0 && yy < R_H && xx >= 0 && xx < R_W)
                    v = in[(size_t)(((b * R_H + yy) * R_W) + xx) * CIN + c];
            }
            av[e] = v;
        }
#pragma unroll
        for (int e = 0; e < 16; ++e) {
            const int k = k0 + hs * 16 + e;
            const int n = n0 + lm;
            _Float16 v = (_Float16)0.f;
            if (k < K && n < Cout) v = wt[(size_t)k * Cout + n];
            bv[e] = v;
        }
        acc = __builtin_amdgcn_wmma_f32_16x16x32_f16(false, av, false, bv,
                                                     (short)0, acc, false, false);
    }
    const int n = n0 + lm;
    if (n >= Cout) return;
    const float bval = bias ? bias[n] : 0.f;
#pragma unroll
    for (int r = 0; r < 8; ++r) {
        const int mm = m0 + hs * 8 + r;
        float v = acc[r] + bval;
        if (act == 1) v = fmaxf(v, 0.f);
        out16[(size_t)mm * outStride + outOff + n] = (_Float16)v;
    }
}

// ---------------------------------------------------------------------------
// Elementwise / glue kernels
// ---------------------------------------------------------------------------
__global__ void k_cast_f16(const float* __restrict__ src, _Float16* __restrict__ dst, int n)
{
    int i = blockIdx.x * blockDim.x + threadIdx.x;
    if (i < n) dst[i] = (_Float16)src[i];
}

// Transpose + pad weights: src [taps*cinr, Cout] f32 -> dst [Cout, taps*cinp] f16
__global__ void k_wt_t(const float* __restrict__ src, _Float16* __restrict__ dst,
                       int Cout, int taps, int cinr, int cinp)
{
    const int Kpad = taps * cinp;
    const int n = Cout * Kpad;
    int i = blockIdx.x * blockDim.x + threadIdx.x;
    if (i >= n) return;
    int nn = i / Kpad, k = i - nn * Kpad;
    int tap = k / cinp, c = k - tap * cinp;
    float v = (c < cinr) ? src[(size_t)(tap * cinr + c) * Cout + nn] : 0.f;
    dst[i] = (_Float16)v;
}

// src [P,C] f32 -> dst f16 at [p*stride + off + c]
__global__ void k_cast_strided(const float* __restrict__ src, _Float16* __restrict__ dst,
                               int C, int stride, int off, int n)
{
    int i = blockIdx.x * blockDim.x + threadIdx.x;
    if (i >= n) return;
    int p = i / C, c = i - p * C;
    dst[(size_t)p * stride + off + c] = (_Float16)src[i];
}

__global__ void k_zero_corr_pad(_Float16* __restrict__ corr16)
{
    int i = blockIdx.x * blockDim.x + threadIdx.x;
    const int PADC = CORR_P - CORR_C;  // 28
    if (i >= R_P * PADC) return;
    int p = i / PADC, c = i - p * PADC;
    corr16[(size_t)p * CORR_P + CORR_C + c] = (_Float16)0.f;
}

__global__ void k_init_coords(float* __restrict__ coords)
{
    int p = blockIdx.x * blockDim.x + threadIdx.x;
    if (p >= R_P) return;
    int rr = p % (R_H * R_W);
    coords[p * 2 + 0] = (float)(rr / R_W);
    coords[p * 2 + 1] = (float)(rr % R_W);
}

// 2x2 avg-pool over the target (h,w) dims per correlation pixel
__global__ void k_pyr(const float* __restrict__ src, float* __restrict__ dst,
                      int hsrc, int wsrc)
{
    const int oh = hsrc >> 1, ow = wsrc >> 1;
    const long long n = (long long)R_P * oh * ow;
    long long i = (long long)blockIdx.x * blockDim.x + threadIdx.x;
    if (i >= n) return;
    int p  = (int)(i / (oh * ow));
    int r  = (int)(i - (long long)p * oh * ow);
    int oy = r / ow, ox = r % ow;
    const float* v = src + (size_t)p * hsrc * wsrc;
    int y0 = oy * 2, x0 = ox * 2;
    float s = v[y0 * wsrc + x0] + v[y0 * wsrc + x0 + 1] +
              v[(y0 + 1) * wsrc + x0] + v[(y0 + 1) * wsrc + x0 + 1];
    dst[i] = 0.25f * s;
}

// bilinear corr lookup over 4 pyramid levels -> corr16 [P, CORR_P] (324 live)
__global__ void k_corr(const int* __restrict__ itersp, int iter,
                       const float* __restrict__ v0, const float* __restrict__ v1,
                       const float* __restrict__ v2, const float* __restrict__ v3,
                       const float* __restrict__ coords, _Float16* __restrict__ corr16)
{
    if (iter >= *itersp) return;
    int i = blockIdx.x * blockDim.x + threadIdx.x;
    if (i >= R_P * CORR_C) return;
    int p = i / CORR_C, j = i - p * CORR_C;
    int l = j / KLU, kk = j - l * KLU;
    int dy = kk / 9 - RADIUS, dx = kk % 9 - RADIUS;
    int h = R_H >> l, w = R_W >> l;
    const float* vol = (l == 0) ? v0 : (l == 1) ? v1 : (l == 2) ? v2 : v3;
    vol += (size_t)p * h * w;
    float inv = 1.f / (float)(1 << l);
    float cy = coords[p * 2 + 0] * inv + (float)dy;
    float cx = coords[p * 2 + 1] * inv + (float)dx;
    float fy0 = floorf(cy), fx0 = floorf(cx);
    float wy = cy - fy0, wx = cx - fx0;
    int y0 = (int)fy0, x0 = (int)fx0;
    float acc = 0.f;
#pragma unroll
    for (int t = 0; t < 4; ++t) {
        int yi = y0 + (t >> 1), xi = x0 + (t & 1);
        float wgt = ((t >> 1) ? wy : 1.f - wy) * ((t & 1) ? wx : 1.f - wx);
        bool valid = (yi >= 0 && yi < h && xi >= 0 && xi < w);
        int yc = min(max(yi, 0), h - 1), xc = min(max(xi, 0), w - 1);
        acc += vol[yc * w + xc] * (valid ? wgt : 0.f);
    }
    corr16[(size_t)p * CORR_P + j] = (_Float16)acc;
}

// flow = coords1 - coords0 -> flow16 [P,2] and x256 channels 254,255
__global__ void k_flow(const int* __restrict__ itersp, int iter,
                       const float* __restrict__ coords,
                       _Float16* __restrict__ flow16, _Float16* __restrict__ x256)
{
    if (iter >= *itersp) return;
    int p = blockIdx.x * blockDim.x + threadIdx.x;
    if (p >= R_P) return;
    int rr = p % (R_H * R_W);
    float fy = coords[p * 2 + 0] - (float)(rr / R_W);
    float fx = coords[p * 2 + 1] - (float)(rr % R_W);
    flow16[p * 2 + 0] = (_Float16)fy;
    flow16[p * 2 + 1] = (_Float16)fx;
    x256[(size_t)p * 256 + 254] = (_Float16)fy;
    x256[(size_t)p * 256 + 255] = (_Float16)fx;
}

// hx[P,384] = concat(net[P,128], x[P,256])
__global__ void k_pack_hx(const int* __restrict__ itersp, int iter,
                          const _Float16* __restrict__ net, const _Float16* __restrict__ x,
                          _Float16* __restrict__ hx)
{
    if (iter >= *itersp) return;
    int i = blockIdx.x * blockDim.x + threadIdx.x;
    if (i >= R_P * 384) return;
    int p = i / 384, c = i - p * 384;
    hx[i] = (c < 128) ? net[(size_t)p * 128 + c] : x[(size_t)p * 256 + (c - 128)];
}

// hx[:, 0:128] = r * net
__global__ void k_rh(const int* __restrict__ itersp, int iter,
                     const _Float16* __restrict__ r, const _Float16* __restrict__ net,
                     _Float16* __restrict__ hx)
{
    if (iter >= *itersp) return;
    int i = blockIdx.x * blockDim.x + threadIdx.x;
    if (i >= R_P * 128) return;
    int p = i / 128, c = i - p * 128;
    float v = (float)r[i] * (float)net[i];
    hx[(size_t)p * 384 + c] = (_Float16)v;
}

// net = (1-z)*net + z*q
__global__ void k_gru_update(const int* __restrict__ itersp, int iter,
                             const _Float16* __restrict__ z, const _Float16* __restrict__ q,
                             _Float16* __restrict__ net)
{
    if (iter >= *itersp) return;
    int i = blockIdx.x * blockDim.x + threadIdx.x;
    if (i >= R_P * 128) return;
    float zz = (float)z[i];
    net[i] = (_Float16)((1.f - zz) * (float)net[i] + zz * (float)q[i]);
}

// flow head conv2 (3x3, 256->2, no act) fused with coords1 += delta (scalar)
__global__ void k_flowhead(const int* __restrict__ itersp, int iter,
                           const _Float16* __restrict__ t256,
                           const float* __restrict__ w, const float* __restrict__ bias,
                           float* __restrict__ coords)
{
    if (iter >= *itersp) return;
    int p = blockIdx.x * blockDim.x + threadIdx.x;
    if (p >= R_P) return;
    int b = p / (R_H * R_W), rr = p % (R_H * R_W);
    int y = rr / R_W, x = rr % R_W;
    float a0 = bias[0], a1 = bias[1];
    for (int ky = 0; ky < 3; ++ky) {
        int yy = y + ky - 1;
        if (yy < 0 || yy >= R_H) continue;
        for (int kx = 0; kx < 3; ++kx) {
            int xx = x + kx - 1;
            if (xx < 0 || xx >= R_W) continue;
            const _Float16* ip = t256 + (size_t)(((b * R_H + yy) * R_W) + xx) * 256;
            const float* wp = w + (size_t)((ky * 3 + kx) * 256) * 2;
            for (int c = 0; c < 256; ++c) {
                float v = (float)ip[c];
                a0 += v * wp[c * 2 + 0];
                a1 += v * wp[c * 2 + 1];
            }
        }
    }
    coords[p * 2 + 0] += a0;
    coords[p * 2 + 1] += a1;
}

// convex upsample: d_out[B, 8H, 8W, 2] with (y,x)->(x,y) flip
__global__ void k_upsample(const int* __restrict__ itersp, int iter,
                           const float* __restrict__ coords, const float* __restrict__ mask,
                           float* __restrict__ out)
{
    if (iter >= *itersp) return;
    int i = blockIdx.x * blockDim.x + threadIdx.x;
    const int UH = 8 * R_H, UW = 8 * R_W;
    if (i >= R_B * UH * UW) return;
    int b  = i / (UH * UW);
    int r  = i - b * (UH * UW);
    int hh = r / UW, ww = r - hh * UW;
    int h = hh >> 3, ii = hh & 7, w = ww >> 3, jj = ww & 7;
    int p = (b * R_H + h) * R_W + w;

    float mv[9], mmax = -1e30f;
#pragma unroll
    for (int k = 0; k < 9; ++k) {
        mv[k] = mask[(size_t)p * 576 + k * 64 + ii * 8 + jj];
        mmax = fmaxf(mmax, mv[k]);
    }
    float s = 0.f;
#pragma unroll
    for (int k = 0; k < 9; ++k) { mv[k] = __expf(mv[k] - mmax); s += mv[k]; }
    float invs = 1.f / s;
    float uy = 0.f, ux = 0.f;
#pragma unroll
    for (int k = 0; k < 9; ++k) {
        int dy = k / 3, dx = k % 3;
        int yy = h + dy - 1, xx = w + dx - 1;
        if (yy < 0 || yy >= R_H || xx < 0 || xx >= R_W) continue;
        int q = (b * R_H + yy) * R_W + xx;
        float wk = mv[k] * invs;
        uy += wk * 8.f * (coords[q * 2 + 0] - (float)yy);
        ux += wk * 8.f * (coords[q * 2 + 1] - (float)xx);
    }
    out[(size_t)i * 2 + 0] = ux;  // flipped: (y,x) -> (x,y)
    out[(size_t)i * 2 + 1] = uy;
}

// ---------------------------------------------------------------------------
// Host orchestration
// ---------------------------------------------------------------------------
extern "C" void kernel_launch(void* const* d_in, const int* in_sizes, int n_in,
                              void* d_out, int out_size, void* d_ws, size_t ws_size,
                              hipStream_t stream)
{
    (void)in_sizes; (void)n_in; (void)out_size; (void)ws_size;
    const float* corr0 = (const float*)d_in[0];
    const float* net0  = (const float*)d_in[1];
    const float* inp0  = (const float*)d_in[2];
    const float* wF[16], *bF[16];
    // order: c1,c2,f1,f2,cf,z1,r1,q1,z2,r2,q2,fh1,fh2,mk1,mk2
    for (int t = 0; t < 15; ++t) { wF[t] = (const float*)d_in[3 + 2 * t]; bF[t] = (const float*)d_in[4 + 2 * t]; }
    const int* itersp = (const int*)d_in[33];
    float* out = (float*)d_out;

    // per-conv geometry: taps, real cin, padded cin, cout
    struct WInfo { int taps, cinr, cinp, cout; };
    const WInfo wi[15] = {
        {1, 324, CORR_P, 256},  // c1 (corr buffer padded to 352)
        {9, 256, 256, 192},     // c2
        {49, 2, 2, 128},        // f1 (scalar path, layout [K,Cout])
        {9, 128, 128, 64},      // f2
        {9, 256, 256, 126},     // cf
        {5, 384, 384, 128}, {5, 384, 384, 128}, {5, 384, 384, 128},   // z1,r1,q1
        {5, 384, 384, 128}, {5, 384, 384, 128}, {5, 384, 384, 128},   // z2,r2,q2
        {9, 128, 128, 256},     // fh1
        {9, 256, 256, 2},       // fh2 (f32 scalar kernel, no f16 weights)
        {9, 128, 128, 256},     // mk1
        {1, 256, 256, 576}      // mk2
    };

    // workspace layout
    char* base = (char*)d_ws;
    size_t off = 0;
    auto take = [&](size_t bytes) -> void* {
        void* p = base + off; off = (off + bytes + 255) & ~(size_t)255; return p;
    };
    float* pyr1 = (float*)take((size_t)R_P * 24 * 32 * 4);
    float* pyr2 = (float*)take((size_t)R_P * 12 * 16 * 4);
    float* pyr3 = (float*)take((size_t)R_P * 6 * 8 * 4);
    _Float16* corr16 = (_Float16*)take((size_t)R_P * CORR_P * 2);
    _Float16* flow16 = (_Float16*)take((size_t)R_P * 2 * 2);
    _Float16* x256   = (_Float16*)take((size_t)R_P * 256 * 2);
    _Float16* cat256 = (_Float16*)take((size_t)R_P * 256 * 2);
    _Float16* t256   = (_Float16*)take((size_t)R_P * 256 * 2);
    _Float16* tf128  = (_Float16*)take((size_t)R_P * 128 * 2);
    _Float16* net16  = (_Float16*)take((size_t)R_P * 128 * 2);
    _Float16* hx384  = (_Float16*)take((size_t)R_P * 384 * 2);
    _Float16* z128   = (_Float16*)take((size_t)R_P * 128 * 2);
    _Float16* r128   = (_Float16*)take((size_t)R_P * 128 * 2);
    _Float16* q128   = (_Float16*)take((size_t)R_P * 128 * 2);
    float*    mask   = (float*)take((size_t)R_P * 576 * 4);
    float*    coords = (float*)take((size_t)R_P * 2 * 4);
    _Float16* wH[15];
    for (int t = 0; t < 15; ++t) {
        if (t == 12) { wH[t] = nullptr; continue; }
        size_t elems = (t == 2) ? (size_t)wi[t].taps * wi[t].cinr * wi[t].cout   // [K,Cout]
                                : (size_t)wi[t].cout * wi[t].taps * wi[t].cinp;  // [Cout,Kpad]
        wH[t] = (_Float16*)take(elems * 2);
    }

    auto g1 = [](long long n) { return dim3((unsigned)((n + 255) / 256)); };

    // ---- setup (runs every call; deterministic) ----
    for (int t = 0; t < 15; ++t) {
        if (t == 12) continue;
        if (t == 2) {
            k_cast_f16<<<g1((long long)wi[t].taps * wi[t].cinr * wi[t].cout), 256, 0, stream>>>(
                wF[t], wH[t], wi[t].taps * wi[t].cinr * wi[t].cout);
        } else {
            long long n = (long long)wi[t].cout * wi[t].taps * wi[t].cinp;
            k_wt_t<<<g1(n), 256, 0, stream>>>(wF[t], wH[t], wi[t].cout, wi[t].taps, wi[t].cinr, wi[t].cinp);
        }
    }
    k_pyr<<<g1((long long)R_P * 24 * 32), 256, 0, stream>>>(corr0, pyr1, 48, 64);
    k_pyr<<<g1((long long)R_P * 12 * 16), 256, 0, stream>>>(pyr1, pyr2, 24, 32);
    k_pyr<<<g1((long long)R_P * 6 * 8),  256, 0, stream>>>(pyr2, pyr3, 12, 16);
    k_zero_corr_pad<<<g1((long long)R_P * (CORR_P - CORR_C)), 256, 0, stream>>>(corr16);
    k_init_coords<<<g1(R_P), 256, 0, stream>>>(coords);
    k_cast_f16<<<g1((long long)R_P * 128), 256, 0, stream>>>(net0, net16, R_P * 128);
    k_cast_strided<<<g1((long long)R_P * 128), 256, 0, stream>>>(inp0, x256, 128, 256, 0, R_P * 128);

    const dim3 blkA(128);
    const dim3 blkS(32);
    const int MT4 = R_P / 64;  // 192 blocks of 4 M-waves
    const int MT  = R_P / 16;  // 768 M-tiles (scalar kernel)

    for (int it = 0; it < 12; ++it) {
        // correlation lookup + flow
        k_corr<<<g1((long long)R_P * CORR_C), 256, 0, stream>>>(itersp, it, corr0, pyr1, pyr2, pyr3, coords, corr16);
        k_flow<<<g1(R_P), 256, 0, stream>>>(itersp, it, coords, flow16, x256);

        // motion encoder
        conv_wmma_a<CORR_P, 1, 1><<<dim3(MT4, 4), blkA, 0, stream>>>(itersp, it, corr16, wH[0], bF[0], t256, nullptr, 256, 256, 0, 1, 1.f);
        conv_wmma_a<256, 3, 3><<<dim3(MT4, 3), blkA, 0, stream>>>(itersp, it, t256, wH[1], bF[1], cat256, nullptr, 192, 256, 0, 1, 1.f);
        conv_wmma_s<2, 7, 7><<<dim3(MT, 8), blkS, 0, stream>>>(itersp, it, flow16, wH[2], bF[2], tf128, 128, 128, 0, 1);
        conv_wmma_a<128, 3, 3><<<dim3(MT4, 1), blkA, 0, stream>>>(itersp, it, tf128, wH[3], bF[3], cat256, nullptr, 64, 256, 192, 1, 1.f);
        conv_wmma_a<256, 3, 3><<<dim3(MT4, 2), blkA, 0, stream>>>(itersp, it, cat256, wH[4], bF[4], x256, nullptr, 126, 256, 128, 1, 1.f);

        // SepConvGRU: horizontal (1x5) then vertical (5x1)
        k_pack_hx<<<g1((long long)R_P * 384), 256, 0, stream>>>(itersp, it, net16, x256, hx384);
        conv_wmma_a<384, 1, 5><<<dim3(MT4, 2), blkA, 0, stream>>>(itersp, it, hx384, wH[5], bF[5], z128, nullptr, 128, 128, 0, 2, 1.f);
        conv_wmma_a<384, 1, 5><<<dim3(MT4, 2), blkA, 0, stream>>>(itersp, it, hx384, wH[6], bF[6], r128, nullptr, 128, 128, 0, 2, 1.f);
        k_rh<<<g1((long long)R_P * 128), 256, 0, stream>>>(itersp, it, r128, net16, hx384);
        conv_wmma_a<384, 1, 5><<<dim3(MT4, 2), blkA, 0, stream>>>(itersp, it, hx384, wH[7], bF[7], q128, nullptr, 128, 128, 0, 3, 1.f);
        k_gru_update<<<g1((long long)R_P * 128), 256, 0, stream>>>(itersp, it, z128, q128, net16);

        k_pack_hx<<<g1((long long)R_P * 384), 256, 0, stream>>>(itersp, it, net16, x256, hx384);
        conv_wmma_a<384, 5, 1><<<dim3(MT4, 2), blkA, 0, stream>>>(itersp, it, hx384, wH[8], bF[8], z128, nullptr, 128, 128, 0, 2, 1.f);
        conv_wmma_a<384, 5, 1><<<dim3(MT4, 2), blkA, 0, stream>>>(itersp, it, hx384, wH[9], bF[9], r128, nullptr, 128, 128, 0, 2, 1.f);
        k_rh<<<g1((long long)R_P * 128), 256, 0, stream>>>(itersp, it, r128, net16, hx384);
        conv_wmma_a<384, 5, 1><<<dim3(MT4, 2), blkA, 0, stream>>>(itersp, it, hx384, wH[10], bF[10], q128, nullptr, 128, 128, 0, 3, 1.f);
        k_gru_update<<<g1((long long)R_P * 128), 256, 0, stream>>>(itersp, it, z128, q128, net16);

        // flow head (delta) + coords update
        conv_wmma_a<128, 3, 3><<<dim3(MT4, 4), blkA, 0, stream>>>(itersp, it, net16, wH[11], bF[11], t256, nullptr, 256, 256, 0, 1, 1.f);
        k_flowhead<<<g1(R_P), 256, 0, stream>>>(itersp, it, t256, wF[12], bF[12], coords);

        // mask head
        conv_wmma_a<128, 3, 3><<<dim3(MT4, 4), blkA, 0, stream>>>(itersp, it, net16, wH[13], bF[13], t256, nullptr, 256, 256, 0, 1, 1.f);
        conv_wmma_a<256, 1, 1><<<dim3(MT4, 9), blkA, 0, stream>>>(itersp, it, t256, wH[14], bF[14], nullptr, mask, 576, 576, 0, 0, 0.25f);

        // convex upsample (last executed iteration leaves final result)
        k_upsample<<<g1((long long)R_B * 8 * R_H * 8 * R_W), 256, 0, stream>>>(itersp, it, coords, mask, out);
    }
}
